// MemResProjections_13477607375124
// MI455X (gfx1250) — matrix operand order repaired and checked by reading, not compiled, not run on hardware
//
#include <hip/hip_runtime.h>
#include <hip/hip_bf16.h>

// MI455X / gfx1250, wave32. All matmuls via v_wmma_f32_16x16x32_bf16.
//
// Pipeline (all on `stream`, workspace in d_ws):
//   1) cast q_w/k_w/gate_w f32->bf16
//   2) prep: RMSNorm(hidden)->vp[0..S), cast memory->vp[S..S+K), cast hidden->hid_bf
//   3) gemm<0,4>: q  = vp[:S] @ qw^T          (bf16 out)
//   4) gemm<0,4>: k  = vp     @ kw^T          (bf16 out)
//   5) gemm<1,4>: alpha = sigmoid(hid @ gw^T + b)   (f32 out)
//   6) gemm<2,2>: logits = q @ k^T * 1/32, causal; fully-masked 64x64 blocks
//                 skip the whole WMMA loop (~47% of QK FLOPs saved)
//   7) softmax rows -> probs (bf16)
//   8) transpose vp -> vpT (so PV GEMM's B operand is contiguous per lane)
//   9) gemm<3,4>: h~ = probs @ vp (via vpT); inner (kv) loop runs only over
//                 [0, rb+64) U [S, S+K) since probs are zero elsewhere (~2x),
//                 then blend with alpha/hidden -> d_out
//
// NT = #16-col tiles per wave in N. NT=4 (wave tile 32x64, 1.5 loads/WMMA)
// for the N=1024 GEMMs; NT=2 for QK since N=2112 = 64*33.

typedef __attribute__((ext_vector_type(16))) __bf16 v16bf;
typedef __attribute__((ext_vector_type(8)))  float  v8f;

#define B_   4
#define S_   2048
#define K_   64
#define H_   1024
#define T_   (S_ + K_)   // 2112

// ---------------- fragment loads (ISA 7.12.2 16-bit layouts) ----------------
// A 16x32: lanes 0-15 row M=lane hold K {0..7, 16..23}; lanes 16-31 same rows
// hold K {8..15, 24..31}.  -> two 16-byte chunks per lane.
__device__ __forceinline__ v16bf load_frag_a(const __bf16* p) {
  union { v16bf v; uint4 u[2]; } t;
  t.u[0] = *reinterpret_cast<const uint4*>(p);       // K = koff .. koff+7
  t.u[1] = *reinterpret_cast<const uint4*>(p + 16);  // K = koff+16 .. koff+23
  return t.v;
}
// B 32x16: lanes 0-15 col N=lane hold K {0..7, 16..23}; lanes 16-31 hold K {8..15, 24..31}.
// With B stored as row-major (N, Hk) this is two 16-byte chunks, matching A.
__device__ __forceinline__ v16bf load_frag_b(const __bf16* p) {
  union { v16bf v; uint4 u[2]; } t;
  t.u[0] = *reinterpret_cast<const uint4*>(p);
  t.u[1] = *reinterpret_cast<const uint4*>(p + 16);
  return t.v;
}

// ---------------- templated WMMA GEMM: C[M,N] = X[M,Hk] @ W[N,Hk]^T ---------
// Block = 128 threads = 4 waves (2 in M x 2 in N). Wave tile = 32 x (16*NT).
// Block tile = 64 x (32*NT). grid = (N/(32*NT), Mrows/64, B). All dims divide
// evenly; all control flow around the WMMAs is workgroup-uniform -> EXEC
// always all-ones.
template <int MODE, int NT>
__global__ __launch_bounds__(128) void gemm_bf16_wmma(
    const __bf16* __restrict__ X, long xbs,
    const __bf16* __restrict__ W, long wbs,
    void* __restrict__ out, long obs, int o_ld,
    int Hk, float scale, int s_local,
    const float* __restrict__ bias,     // MODE 1
    const float* __restrict__ alpha,    // MODE 3
    const float* __restrict__ hidden,   // MODE 3
    long fbs)                           // MODE 3: alpha/hidden batch stride
{
  const int b    = blockIdx.z;
  const int tid  = threadIdx.x;
  const int lane = tid & 31;
  const int wid  = tid >> 5;
  const int wm   = wid >> 1;                 // 0..1
  const int wn   = wid & 1;                  // 0..1
  const int rb   = blockIdx.y * 64;          // block row base (within batch)
  const int cb   = blockIdx.x * (32 * NT);   // block col base
  const int m0   = rb + wm * 32;
  const int n0   = cb + wn * (16 * NT);
  const int hl   = lane >> 4;                // half-wave select

  // QK^T: a block that lies fully above the diagonal in the local region is
  // entirely masked -> skip the GEMM, emit -1e30, done.
  if (MODE == 2 && cb < s_local && cb >= rb + 64) {
    float* O = (float*)out + (long)b * obs;
#pragma unroll
    for (int mt = 0; mt < 2; ++mt)
#pragma unroll
      for (int nt = 0; nt < NT; ++nt)
#pragma unroll
        for (int r = 0; r < 8; ++r) {
          const int row = m0 + mt * 16 + r + (hl << 3);
          const int col = n0 + nt * 16 + (lane & 15);
          O[(long)row * o_ld + col] = -1e30f;
        }
    return;
  }

  const __bf16* Xb = X + (long)b * xbs;
  const __bf16* Wb = W + (long)b * wbs;
  const __bf16* a0p = Xb + (long)(m0 + (lane & 15)) * Hk + hl * 8;
  const __bf16* a1p = a0p + (long)16 * Hk;
  const __bf16* bp[NT];
#pragma unroll
  for (int nt = 0; nt < NT; ++nt)
    bp[nt] = Wb + (long)(n0 + nt * 16 + (lane & 15)) * Hk + hl * 8;

  v8f acc[2][NT] = {};

  auto step = [&]() {
    v16bf a0 = load_frag_a(a0p);
    v16bf a1 = load_frag_a(a1p);
    v16bf bfr[NT];
#pragma unroll
    for (int nt = 0; nt < NT; ++nt) bfr[nt] = load_frag_b(bp[nt]);
#pragma unroll
    for (int nt = 0; nt < NT; ++nt) {
      acc[0][nt] = __builtin_amdgcn_wmma_f32_16x16x32_bf16(false, a0, false, bfr[nt], (short)0, acc[0][nt], false, false);
      acc[1][nt] = __builtin_amdgcn_wmma_f32_16x16x32_bf16(false, a1, false, bfr[nt], (short)0, acc[1][nt], false, false);
    }
    a0p += 32; a1p += 32;
#pragma unroll
    for (int nt = 0; nt < NT; ++nt) bp[nt] += 32;
  };

  // PV: probs are exactly zero for kv in [rb+64, S) due to causality, so the
  // reduction only needs [0, rb+64) and the memory block [S, T). Bounds are
  // blockIdx-uniform.
  const int seg0_end = (MODE == 3) ? (rb + 64) : Hk;

#pragma unroll 2
  for (int k = 0; k < seg0_end; k += 32) step();

  if (MODE == 3) {
    const int delta = S_ - seg0_end;   // jump to the memory rows
    a0p += delta; a1p += delta;
#pragma unroll
    for (int nt = 0; nt < NT; ++nt) bp[nt] += delta;
#pragma unroll 2
    for (int k = S_; k < Hk; k += 32) step();
  }

  // C/D layout: VGPR r, lanes 0-15 -> M = r, lanes 16-31 -> M = r+8; N = lane&15
#pragma unroll
  for (int mt = 0; mt < 2; ++mt)
#pragma unroll
    for (int nt = 0; nt < NT; ++nt)
#pragma unroll
      for (int r = 0; r < 8; ++r) {
        const int row = m0 + mt * 16 + r + (hl << 3);
        const int col = n0 + nt * 16 + (lane & 15);
        const float v = acc[mt][nt][r];
        const long  idx = (long)row * o_ld + col;
        if (MODE == 0) {               // bf16 out (q / k projections)
          ((__bf16*)out)[(long)b * obs + idx] = (__bf16)v;
        } else if (MODE == 1) {        // gate: sigmoid(acc + bias)
          const float g = v + bias[col];
          ((float*)out)[(long)b * obs + idx] = 1.0f / (1.0f + __expf(-g));
        } else if (MODE == 2) {        // QK^T: scale + causal mask on local cols
          float l = v * scale;
          if (col < s_local && col > row) l = -1e30f;
          ((float*)out)[(long)b * obs + idx] = l;
        } else {                       // PV + gate blend -> final output
          const float a  = alpha[(long)b * fbs + idx];
          const float h  = hidden[(long)b * fbs + idx];
          ((float*)out)[(long)b * obs + idx] = (1.0f - a) * h + a * v;
        }
      }
}

// ---------------- helpers ----------------
__global__ __launch_bounds__(256) void cast_f32_bf16(const float* __restrict__ in,
                                                     __bf16* __restrict__ out, int n) {
  int i = blockIdx.x * 256 + threadIdx.x;
  if (i < n) out[i] = (__bf16)in[i];
}

// one block (256 thr) per row of vp; RMSNorm for local rows, cast for memory rows
__global__ __launch_bounds__(256) void prep_kernel(
    const float* __restrict__ hidden, const float* __restrict__ memory,
    const float* __restrict__ norm_w, __bf16* __restrict__ vp,
    __bf16* __restrict__ hid_bf) {
  __shared__ float red[256];
  const int r = blockIdx.x;
  const int b = r / T_;
  const int t = r - b * T_;
  const int tid = threadIdx.x;
  if (t < S_) {
    const float* x = hidden + ((long)b * S_ + t) * H_;
    const float4 v = *reinterpret_cast<const float4*>(x + tid * 4);
    float ss = v.x * v.x + v.y * v.y + v.z * v.z + v.w * v.w;
    red[tid] = ss;
    __syncthreads();
    for (int s = 128; s > 0; s >>= 1) {
      if (tid < s) red[tid] += red[tid + s];
      __syncthreads();
    }
    const float rms = __frsqrt_rn(red[0] * (1.0f / H_) + 1e-6f);
    __bf16* vo = vp + (long)r * H_;
    __bf16* ho = hid_bf + ((long)b * S_ + t) * H_;
    const float w0 = norm_w[tid * 4 + 0], w1 = norm_w[tid * 4 + 1];
    const float w2 = norm_w[tid * 4 + 2], w3 = norm_w[tid * 4 + 3];
    vo[tid * 4 + 0] = (__bf16)(v.x * rms * w0);
    vo[tid * 4 + 1] = (__bf16)(v.y * rms * w1);
    vo[tid * 4 + 2] = (__bf16)(v.z * rms * w2);
    vo[tid * 4 + 3] = (__bf16)(v.w * rms * w3);
    ho[tid * 4 + 0] = (__bf16)v.x;
    ho[tid * 4 + 1] = (__bf16)v.y;
    ho[tid * 4 + 2] = (__bf16)v.z;
    ho[tid * 4 + 3] = (__bf16)v.w;
  } else {
    const float* m = memory + ((long)b * K_ + (t - S_)) * H_;
    const float4 v = *reinterpret_cast<const float4*>(m + tid * 4);
    __bf16* vo = vp + (long)r * H_;
    vo[tid * 4 + 0] = (__bf16)v.x;
    vo[tid * 4 + 1] = (__bf16)v.y;
    vo[tid * 4 + 2] = (__bf16)v.z;
    vo[tid * 4 + 3] = (__bf16)v.w;
  }
}

// one block (256 thr) per logits row of length T_=2112; masked cols are -1e30
__global__ __launch_bounds__(256) void softmax_kernel(
    const float* __restrict__ logits, __bf16* __restrict__ probs) {
  __shared__ float red[256];
  const long base = (long)blockIdx.x * T_;
  const int tid = threadIdx.x;
  float vals[9];
  float mx = -3.0e38f;
#pragma unroll
  for (int i = 0; i < 9; ++i) {
    const int t = tid + i * 256;
    vals[i] = (t < T_) ? logits[base + t] : -3.0e38f;
    mx = fmaxf(mx, vals[i]);
  }
  red[tid] = mx;
  __syncthreads();
  for (int s = 128; s > 0; s >>= 1) {
    if (tid < s) red[tid] = fmaxf(red[tid], red[tid + s]);
    __syncthreads();
  }
  mx = red[0];
  __syncthreads();
  float ex[9], sum = 0.f;
#pragma unroll
  for (int i = 0; i < 9; ++i) {
    const int t = tid + i * 256;
    ex[i] = (t < T_) ? __expf(vals[i] - mx) : 0.f;
    sum += ex[i];
  }
  red[tid] = sum;
  __syncthreads();
  for (int s = 128; s > 0; s >>= 1) {
    if (tid < s) red[tid] += red[tid + s];
    __syncthreads();
  }
  const float inv = 1.0f / red[0];
#pragma unroll
  for (int i = 0; i < 9; ++i) {
    const int t = tid + i * 256;
    if (t < T_) probs[base + t] = (__bf16)(ex[i] * inv);
  }
}

// vp (B, T_, H_) -> vpT (B, H_, T_), 32x32 LDS tiles
__global__ __launch_bounds__(256) void transpose_kernel(
    const __bf16* __restrict__ vp, __bf16* __restrict__ vpT) {
  __shared__ __bf16 tile[32][33];
  const int b = blockIdx.z;
  const int h0 = blockIdx.x * 32;
  const int t0 = blockIdx.y * 32;
  const int tx = threadIdx.x, ty = threadIdx.y;
  const __bf16* src = vp + (long)b * T_ * H_;
  __bf16* dst = vpT + (long)b * H_ * T_;
#pragma unroll
  for (int j = 0; j < 4; ++j)
    tile[ty + j * 8][tx] = src[(long)(t0 + ty + j * 8) * H_ + h0 + tx];
  __syncthreads();
#pragma unroll
  for (int j = 0; j < 4; ++j)
    dst[(long)(h0 + ty + j * 8) * T_ + t0 + tx] = tile[tx][ty + j * 8];
}

// ---------------- host ----------------
extern "C" void kernel_launch(void* const* d_in, const int* in_sizes, int n_in,
                              void* d_out, int out_size, void* d_ws, size_t ws_size,
                              hipStream_t stream) {
  (void)in_sizes; (void)n_in; (void)out_size; (void)ws_size;
  const float* hidden = (const float*)d_in[0];
  const float* memory = (const float*)d_in[1];
  const float* q_w    = (const float*)d_in[2];
  const float* k_w    = (const float*)d_in[3];
  const float* norm_w = (const float*)d_in[4];
  const float* gate_w = (const float*)d_in[5];
  const float* gate_b = (const float*)d_in[6];

  char* ws = (char*)d_ws;
  size_t off = 0;
  auto alloc = [&](size_t bytes) { size_t r = off; off = (off + bytes + 255) & ~(size_t)255; return r; };
  __bf16* qw_bf  = (__bf16*)(ws + alloc((size_t)2 * H_ * H_));
  __bf16* kw_bf  = (__bf16*)(ws + alloc((size_t)2 * H_ * H_));
  __bf16* gw_bf  = (__bf16*)(ws + alloc((size_t)2 * H_ * H_));
  __bf16* hid_bf = (__bf16*)(ws + alloc((size_t)2 * B_ * S_ * H_));
  __bf16* vp     = (__bf16*)(ws + alloc((size_t)2 * B_ * T_ * H_));
  __bf16* q_bf   = (__bf16*)(ws + alloc((size_t)2 * B_ * S_ * H_));
  __bf16* k_bf   = (__bf16*)(ws + alloc((size_t)2 * B_ * T_ * H_));
  __bf16* vpT    = (__bf16*)(ws + alloc((size_t)2 * B_ * H_ * T_));
  float*  alpha  = (float*) (ws + alloc((size_t)4 * B_ * S_ * H_));
  float*  logits = (float*) (ws + alloc((size_t)4 * B_ * S_ * T_));
  __bf16* probs  = (__bf16*)(ws + alloc((size_t)2 * B_ * S_ * T_));

  // 1) weight casts
  cast_f32_bf16<<<(H_ * H_ + 255) / 256, 256, 0, stream>>>(q_w, qw_bf, H_ * H_);
  cast_f32_bf16<<<(H_ * H_ + 255) / 256, 256, 0, stream>>>(k_w, kw_bf, H_ * H_);
  cast_f32_bf16<<<(H_ * H_ + 255) / 256, 256, 0, stream>>>(gate_w, gw_bf, H_ * H_);

  // 2) RMSNorm + casts -> vp, hid_bf
  prep_kernel<<<B_ * T_, 256, 0, stream>>>(hidden, memory, norm_w, vp, hid_bf);

  // 3) q = vp[:S] @ qw^T   (bf16)
  gemm_bf16_wmma<0, 4><<<dim3(H_ / 128, S_ / 64, B_), 128, 0, stream>>>(
      vp, (long)T_ * H_, qw_bf, 0, q_bf, (long)S_ * H_, H_, H_, 1.f, 0,
      nullptr, nullptr, nullptr, 0);

  // 4) k = vp @ kw^T   (bf16)
  gemm_bf16_wmma<0, 4><<<dim3(H_ / 128, T_ / 64, B_), 128, 0, stream>>>(
      vp, (long)T_ * H_, kw_bf, 0, k_bf, (long)T_ * H_, H_, H_, 1.f, 0,
      nullptr, nullptr, nullptr, 0);

  // 5) alpha = sigmoid(hid @ gw^T + b)
  gemm_bf16_wmma<1, 4><<<dim3(H_ / 128, S_ / 64, B_), 128, 0, stream>>>(
      hid_bf, (long)S_ * H_, gw_bf, 0, alpha, (long)S_ * H_, H_, H_, 1.f, 0,
      gate_b, nullptr, nullptr, 0);

  // 6) logits = q @ k^T * 1/sqrt(H), causal; masked blocks skip the GEMM
  gemm_bf16_wmma<2, 2><<<dim3(T_ / 64, S_ / 64, B_), 128, 0, stream>>>(
      q_bf, (long)S_ * H_, k_bf, (long)T_ * H_, logits, (long)S_ * T_, T_, H_,
      0.03125f, S_, nullptr, nullptr, nullptr, 0);

  // 7) softmax rows -> probs bf16
  softmax_kernel<<<B_ * S_, 256, 0, stream>>>(logits, probs);

  // 8) vp -> vpT
  transpose_kernel<<<dim3(H_ / 32, T_ / 32, B_), dim3(32, 8), 0, stream>>>(vp, vpT);

  // 9) h~ = probs @ vp (via vpT as W), causal-truncated kv loop, blend -> out
  gemm_bf16_wmma<3, 4><<<dim3(H_ / 128, S_ / 64, B_), 128, 0, stream>>>(
      probs, (long)S_ * T_, vpT, (long)H_ * T_, d_out, (long)S_ * H_, H_, T_,
      1.f, 0, nullptr, alpha, hidden, (long)S_ * H_);
}
